// RDB_56934086476492
// MI455X (gfx1250) — matrix operand order repaired and checked by reading, not compiled, not run on hardware
//
#include <hip/hip_runtime.h>
#include <hip/hip_bf16.h>

typedef __attribute__((ext_vector_type(16))) __bf16 v16bf;
typedef __attribute__((ext_vector_type(8)))  float  v8f;

typedef unsigned short ushort_t;
typedef unsigned int   uint_t;

#define BN 16
#define HH 96
#define WW 96
#define HP 98                  // padded spatial dim (1-px halo)
#define CTOT 320
#define NPOS (BN*HH*WW)        // 147456
#define HWPOS (HH*WW)          // 9216

union AVec {
    v16bf v;
    uint4 q[2];
};

__device__ __forceinline__ ushort_t f2bf(float f) {
    uint_t u = __float_as_uint(f);
    u += 0x7FFFu + ((u >> 16) & 1u);   // round-to-nearest-even
    return (ushort_t)(u >> 16);
}
__device__ __forceinline__ float bf2f(ushort_t h) {
    return __uint_as_float(((uint_t)h) << 16);
}

__device__ __forceinline__ v8f wmma_bf16(const AVec& a, const AVec& b, v8f c) {
    return __builtin_amdgcn_wmma_f32_16x16x32_bf16(false, a.v, false, b.v,
                                                   (short)0, c, false, false);
}

// ---------------------------------------------------------------------------
// 0) zero the padded NHWC activation buffer (halo must be 0 for all channels)
// ---------------------------------------------------------------------------
__global__ void k_zero_act(uint4* __restrict__ p, size_t n16) {
    size_t i = (size_t)blockIdx.x * blockDim.x + threadIdx.x;
    if (i < n16) p[i] = make_uint4(0u, 0u, 0u, 0u);
}

__global__ void k_zero(float* __restrict__ p, int n) {
    int i = blockIdx.x * blockDim.x + threadIdx.x;
    if (i < n) p[i] = 0.f;
}

// ---------------------------------------------------------------------------
// 1) x (NCHW fp32, 64 ch) -> act (padded NHWC bf16, channels 0..63 of 320)
// ---------------------------------------------------------------------------
__global__ void k_convert_x(const float* __restrict__ x, ushort_t* __restrict__ act) {
    size_t id = (size_t)blockIdx.x * blockDim.x + threadIdx.x;
    if (id >= (size_t)BN * HWPOS * 64) return;
    int ch = (int)(id & 63);
    size_t r = id >> 6;                 // b*HWPOS + p
    int b = (int)(r / HWPOS);
    int p = (int)(r - (size_t)b * HWPOS);
    int y = p / WW;
    int xx = p - y * WW;
    float v = x[((size_t)(b * 64 + ch)) * HWPOS + p];
    act[((size_t)(b * HP + y + 1) * HP + (xx + 1)) * CTOT + ch] = f2bf(v);
}

// ---------------------------------------------------------------------------
// 2) conv weights [32][cin][3][3] fp32 -> Wt bf16 [32][tap=9][cin] (all layers)
// ---------------------------------------------------------------------------
__global__ void k_convert_w(const float* w0, const float* w1, const float* w2,
                            const float* w3, const float* w4, const float* w5,
                            const float* w6, const float* w7,
                            ushort_t* __restrict__ wt, int totalElems) {
    int e = blockIdx.x * blockDim.x + threadIdx.x;
    if (e >= totalElems) return;
    const float* ws[8] = {w0, w1, w2, w3, w4, w5, w6, w7};
    int rem = e;
    size_t dstBase = 0;
    for (int c = 0; c < 8; ++c) {
        int cin = 64 + 32 * c;
        int sz  = 32 * 9 * cin;
        if (rem < sz) {
            int m   = rem / (9 * cin);
            int r2  = rem - m * 9 * cin;
            int tap = r2 / cin;
            int ch  = r2 - tap * cin;
            float v = ws[c][((size_t)(m * cin + ch)) * 9 + tap];
            wt[dstBase + (size_t)rem] = f2bf(v);
            return;
        }
        rem -= sz;
        dstBase += (size_t)sz;
    }
}

// lff_w [64][320] fp32 -> bf16
__global__ void k_convert_lff(const float* __restrict__ lw, ushort_t* __restrict__ wl) {
    int e = blockIdx.x * blockDim.x + threadIdx.x;
    if (e < 64 * CTOT) wl[e] = f2bf(lw[e]);
}

// ---------------------------------------------------------------------------
// 3) conv layer as implicit GEMM (CIN is a compile-time constant).
//    M=32, K=9*CIN, N=147456. Block = 8 waves; each wave: 32(M) x 32(N).
//    Halo-padded NHWC input + pure 32-bit index addressing =>
//    branch/division-free inner loop: 4 global_load_b128 (B, saddr form)
//    + 4 ds_load_b128 (A) + 4 v_wmma per step.
// ---------------------------------------------------------------------------
template<int CIN>
__global__ __launch_bounds__(256) void k_conv_gemm(ushort_t* act,
                                                   const ushort_t* __restrict__ wt,
                                                   const float* __restrict__ bias,
                                                   int chbase) {
    constexpr int K9 = 9 * CIN;
    extern __shared__ ushort_t lds_w[];      // 32 * K9 halfs

    // cooperative stage of the 32 x K9 weight tile (dword copies)
    {
        const uint_t* src = (const uint_t*)wt;
        uint_t* dst = (uint_t*)lds_w;
        for (int i = threadIdx.x; i < (32 * K9) >> 1; i += 256) dst[i] = src[i];
    }
    __syncthreads();

    const int tid  = threadIdx.x;
    const int wave = tid >> 5;
    const int lane = tid & 31;
    const int n    = lane & 15;
    const int hi   = lane >> 4;

    const int posBase = blockIdx.x * 256 + wave * 32;

    // per-subtile center indices into the padded NHWC buffer (32-bit, in halfs)
    const int pos0 = posBase + n;
    const int pos1 = pos0 + 16;
    const int b0s  = pos0 / HWPOS;
    const int rp0  = pos0 - b0s * HWPOS;
    const int y0   = rp0 / WW;
    const int x0   = rp0 - y0 * WW;
    const int b1s  = pos1 / HWPOS;
    const int rp1  = pos1 - b1s * HWPOS;
    const int y1   = rp1 / WW;
    const int x1   = rp1 - y1 * WW;
    const int ctr0 = ((b0s * HP + y0 + 1) * HP + (x0 + 1)) * CTOT;
    const int ctr1 = ((b1s * HP + y1 + 1) * HP + (x1 + 1)) * CTOT;

    v8f acc00 = {}, acc10 = {}, acc01 = {}, acc11 = {};
    const int arow0 = n * K9 + hi * 8;
    const int arow1 = (n + 16) * K9 + hi * 8;

#pragma unroll
    for (int tap = 0; tap < 9; ++tap) {
        const int dy  = tap / 3 - 1;
        const int dx  = tap - (tap / 3) * 3 - 1;
        const int off = (dy * HP + dx) * CTOT + hi * 8;
        const int bi0 = ctr0 + off;
        const int bi1 = ctr1 + off;
        const int ai0 = arow0 + tap * CIN;
        const int ai1 = arow1 + tap * CIN;
        __builtin_prefetch(act + bi0 + CIN, 0, 1);     // global_prefetch_b8 hint
#pragma unroll 2
        for (int c0 = 0; c0 < CIN; c0 += 32) {
            AVec bv0, bv1, av0, av1;
            bv0.q[0] = *(const uint4*)(act + bi0 + c0);
            bv0.q[1] = *(const uint4*)(act + bi0 + c0 + 16);
            bv1.q[0] = *(const uint4*)(act + bi1 + c0);
            bv1.q[1] = *(const uint4*)(act + bi1 + c0 + 16);
            av0.q[0] = *(const uint4*)(lds_w + ai0 + c0);
            av0.q[1] = *(const uint4*)(lds_w + ai0 + c0 + 16);
            av1.q[0] = *(const uint4*)(lds_w + ai1 + c0);
            av1.q[1] = *(const uint4*)(lds_w + ai1 + c0 + 16);

            acc00 = wmma_bf16(av0, bv0, acc00);
            acc10 = wmma_bf16(av1, bv0, acc10);
            acc01 = wmma_bf16(av0, bv1, acc01);
            acc11 = wmma_bf16(av1, bv1, acc11);
        }
    }

    // epilogue: bias + relu, 8 consecutive bf16 channels per lane, b128 store
#pragma unroll
    for (int sub = 0; sub < 2; ++sub) {
        const int ctr = sub ? ctr1 : ctr0;
#pragma unroll
        for (int t = 0; t < 2; ++t) {
            const v8f& acc = sub ? (t ? acc11 : acc01) : (t ? acc10 : acc00);
            const int mb = t * 16 + hi * 8;
            uint_t pk[4];
#pragma unroll
            for (int r = 0; r < 4; ++r) {
                float v0 = fmaxf(acc[2 * r]     + bias[mb + 2 * r],     0.f);
                float v1 = fmaxf(acc[2 * r + 1] + bias[mb + 2 * r + 1], 0.f);
                pk[r] = (uint_t)f2bf(v0) | ((uint_t)f2bf(v1) << 16);
            }
            *(uint4*)(act + ctr + chbase + mb) = make_uint4(pk[0], pk[1], pk[2], pk[3]);
        }
    }
}

// ---------------------------------------------------------------------------
// 4) global average pool (sum via atomics; mean applied in MLP kernel)
// ---------------------------------------------------------------------------
__global__ void k_pool(const ushort_t* __restrict__ act, float* __restrict__ pooled) {
    int b = blockIdx.x / HH;
    int y = blockIdx.x - b * HH;
    int c = threadIdx.x;
    float s = 0.f;
    const int base = ((b * HP + y + 1) * HP + 1) * CTOT + c;
    for (int x = 0; x < WW; ++x) s += bf2f(act[base + x * CTOT]);
    atomicAdd(&pooled[b * CTOT + c], s);
}

// ---------------------------------------------------------------------------
// 5) gating MLP
// ---------------------------------------------------------------------------
__global__ void k_mlp(const float* __restrict__ pooled, const float* __restrict__ s,
                      const float* __restrict__ w1, const float* __restrict__ b1,
                      const float* __restrict__ w2, const float* __restrict__ b2,
                      float* __restrict__ scale) {
    __shared__ float pr[CTOT + 1];
    __shared__ float h[16];
    int b = blockIdx.x;
    int t = threadIdx.x;
    for (int i = t; i < CTOT; i += 64) pr[i] = pooled[b * CTOT + i] * (1.f / (float)HWPOS);
    if (t == 0) pr[CTOT] = s[0];
    __syncthreads();
    if (t < 16) {
        float a = b1[t];
        for (int i = 0; i <= CTOT; ++i) a += pr[i] * w1[i * 16 + t];
        h[t] = fmaxf(a, 0.f);
    }
    __syncthreads();
    float a = b2[t];
    for (int j = 0; j < 16; ++j) a += h[j] * w2[j * 64 + t];
    scale[b * 64 + t] = 1.f / (1.f + __expf(-a));
}

// ---------------------------------------------------------------------------
// 6) gated 1x1 conv + bias + residual (WMMA GEMM, M=64, K=320).
//    Block = 8 waves: 2(M) x 4(N); each wave 32(M) x 32(N).
// ---------------------------------------------------------------------------
__global__ __launch_bounds__(256) void k_lff_gemm(const ushort_t* __restrict__ act,
                                                  const ushort_t* __restrict__ wl,
                                                  const float* __restrict__ lffb,
                                                  const float* __restrict__ scale,
                                                  const float* __restrict__ xin,
                                                  float* __restrict__ out) {
    extern __shared__ ushort_t lds_w[];      // 64 * 320 halfs
    {
        const uint_t* src = (const uint_t*)wl;
        uint_t* dst = (uint_t*)lds_w;
        for (int i = threadIdx.x; i < (64 * CTOT) >> 1; i += 256) dst[i] = src[i];
    }
    __syncthreads();

    const int tid   = threadIdx.x;
    const int wave  = tid >> 5;
    const int lane  = tid & 31;
    const int n     = lane & 15;
    const int hi    = lane >> 4;
    const int nsub  = wave & 3;
    const int mbase = (wave >> 2) * 32;

    const int posBase = blockIdx.x * 128 + nsub * 32;
    const int pos0 = posBase + n;
    const int pos1 = pos0 + 16;
    const int b0s  = pos0 / HWPOS;
    const int rp0  = pos0 - b0s * HWPOS;
    const int y0   = rp0 / WW;
    const int x0   = rp0 - y0 * WW;
    const int b1s  = pos1 / HWPOS;
    const int rp1  = pos1 - b1s * HWPOS;
    const int y1   = rp1 / WW;
    const int x1   = rp1 - y1 * WW;
    const int bi0  = ((b0s * HP + y0 + 1) * HP + (x0 + 1)) * CTOT + hi * 8;
    const int bi1  = ((b1s * HP + y1 + 1) * HP + (x1 + 1)) * CTOT + hi * 8;

    v8f acc00 = {}, acc10 = {}, acc01 = {}, acc11 = {};
    const int arow0 = (mbase + n) * CTOT + hi * 8;
    const int arow1 = (mbase + n + 16) * CTOT + hi * 8;

#pragma unroll 2
    for (int k0 = 0; k0 < CTOT; k0 += 32) {
        AVec bv0, bv1, av0, av1;
        bv0.q[0] = *(const uint4*)(act + bi0 + k0);
        bv0.q[1] = *(const uint4*)(act + bi0 + k0 + 16);
        bv1.q[0] = *(const uint4*)(act + bi1 + k0);
        bv1.q[1] = *(const uint4*)(act + bi1 + k0 + 16);
        av0.q[0] = *(const uint4*)(lds_w + arow0 + k0);
        av0.q[1] = *(const uint4*)(lds_w + arow0 + k0 + 16);
        av1.q[0] = *(const uint4*)(lds_w + arow1 + k0);
        av1.q[1] = *(const uint4*)(lds_w + arow1 + k0 + 16);

        acc00 = wmma_bf16(av0, bv0, acc00);
        acc10 = wmma_bf16(av1, bv0, acc10);
        acc01 = wmma_bf16(av0, bv1, acc01);
        acc11 = wmma_bf16(av1, bv1, acc11);
    }

#pragma unroll
    for (int sub = 0; sub < 2; ++sub) {
        const int bb = sub ? b1s : b0s;
        const int rp = sub ? rp1 : rp0;
#pragma unroll
        for (int t = 0; t < 2; ++t) {
            const v8f& acc = sub ? (t ? acc11 : acc01) : (t ? acc10 : acc00);
#pragma unroll
            for (int r = 0; r < 8; ++r) {
                int o = mbase + t * 16 + hi * 8 + r;
                int idx = (bb * 64 + o) * HWPOS + rp;
                out[idx] = acc[r] * scale[bb * 64 + o] + lffb[o] + xin[idx];
            }
        }
    }
}

// ---------------------------------------------------------------------------
// launch
// ---------------------------------------------------------------------------
extern "C" void kernel_launch(void* const* d_in, const int* in_sizes, int n_in,
                              void* d_out, int out_size, void* d_ws, size_t ws_size,
                              hipStream_t stream) {
    (void)in_sizes; (void)n_in; (void)out_size; (void)ws_size;

    const float* x    = (const float*)d_in[0];
    const float* s    = (const float*)d_in[1];
    const float* cw[8];
    const float* cb[8];
    for (int c = 0; c < 8; ++c) { cw[c] = (const float*)d_in[2 + c]; cb[c] = (const float*)d_in[10 + c]; }
    const float* lffw = (const float*)d_in[18];
    const float* lffb = (const float*)d_in[19];
    const float* w1   = (const float*)d_in[20];
    const float* b1   = (const float*)d_in[21];
    const float* w2   = (const float*)d_in[22];
    const float* b2   = (const float*)d_in[23];
    float* out = (float*)d_out;

    // workspace layout
    char* ws = (char*)d_ws;
    size_t actElems = (size_t)BN * HP * HP * CTOT;              // padded NHWC halfs
    size_t actBytes = actElems * sizeof(ushort_t);              // ~98.3 MB
    int wtElems = 0;
    for (int c = 0; c < 8; ++c) wtElems += 32 * 9 * (64 + 32 * c); // 405504
    ushort_t* act    = (ushort_t*)(ws);
    ushort_t* wt     = (ushort_t*)(ws + actBytes);
    ushort_t* wl     = (ushort_t*)(ws + actBytes + (size_t)wtElems * 2);
    float*    pooled = (float*)   (ws + actBytes + (size_t)wtElems * 2 + (size_t)64 * CTOT * 2);
    float*    scale  = (float*)((char*)pooled + (size_t)BN * CTOT * 4);

    // prep: clear padded activations (halo correctness) + pooled accumulators
    {
        size_t n16 = actElems / 8;    // uint4 chunks
        k_zero_act<<<(unsigned)((n16 + 255) / 256), 256, 0, stream>>>((uint4*)act, n16);
    }
    k_zero<<<(BN * CTOT + 255) / 256, 256, 0, stream>>>(pooled, BN * CTOT);
    {
        size_t nx = (size_t)BN * HWPOS * 64;
        k_convert_x<<<(unsigned)((nx + 255) / 256), 256, 0, stream>>>(x, act);
    }
    k_convert_w<<<(wtElems + 255) / 256, 256, 0, stream>>>(cw[0], cw[1], cw[2], cw[3],
                                                           cw[4], cw[5], cw[6], cw[7],
                                                           wt, wtElems);
    k_convert_lff<<<(64 * CTOT + 255) / 256, 256, 0, stream>>>(lffw, wl);

    // dense conv chain (templated WMMA implicit GEMM per layer)
    size_t wOff = 0;
    const int nblk = NPOS / 256;
    for (int c = 0; c < 8; ++c) {
        int cin = 64 + 32 * c;
        size_t shmem = (size_t)32 * 9 * cin * sizeof(ushort_t);  // <= 165,888 B
        int chbase = 64 + 32 * c;
        switch (c) {
        case 0: k_conv_gemm< 64><<<nblk, 256, shmem, stream>>>(act, wt + wOff, cb[c], chbase); break;
        case 1: k_conv_gemm< 96><<<nblk, 256, shmem, stream>>>(act, wt + wOff, cb[c], chbase); break;
        case 2: k_conv_gemm<128><<<nblk, 256, shmem, stream>>>(act, wt + wOff, cb[c], chbase); break;
        case 3: k_conv_gemm<160><<<nblk, 256, shmem, stream>>>(act, wt + wOff, cb[c], chbase); break;
        case 4: k_conv_gemm<192><<<nblk, 256, shmem, stream>>>(act, wt + wOff, cb[c], chbase); break;
        case 5: k_conv_gemm<224><<<nblk, 256, shmem, stream>>>(act, wt + wOff, cb[c], chbase); break;
        case 6: k_conv_gemm<256><<<nblk, 256, shmem, stream>>>(act, wt + wOff, cb[c], chbase); break;
        case 7: k_conv_gemm<288><<<nblk, 256, shmem, stream>>>(act, wt + wOff, cb[c], chbase); break;
        }
        wOff += (size_t)32 * 9 * cin;
    }

    // pool -> MLP gates
    k_pool<<<BN * HH, CTOT, 0, stream>>>(act, pooled);
    k_mlp<<<BN, 64, 0, stream>>>(pooled, s, w1, b1, w2, b2, scale);

    // gated 1x1 + residual (WMMA GEMM)
    k_lff_gemm<<<NPOS / 128, 256, (size_t)64 * CTOT * sizeof(ushort_t), stream>>>(
        act, wl, lffb, scale, x, out);
}
